// SchNetCutoffInteraction_16234976379044
// MI455X (gfx1250) — compile-verified, matrix operand order checked by reading, not run on hardware
//
#include <hip/hip_runtime.h>
#include <hip/hip_bf16.h>

// ---------------------------------------------------------------------------
// SchNet cfconv interaction for MI455X (gfx1250, wave32, WMMA).
// bf16 WMMA (f32 accumulate) for all GEMMs; f32 transcendental epilogues via
// hardware v_exp/v_log; fused filter-net + cutoff + gather + masked reduce so
// the 268 MB filter tensor never exists. LDS staging uses CDNA5
// GLOBAL_LOAD_ASYNC_TO_LDS_B128 (ASYNCcnt) when the builtin is available.
// ---------------------------------------------------------------------------

typedef __attribute__((ext_vector_type(16))) __bf16 bf16x16;
typedef __attribute__((ext_vector_type(8)))  __bf16 bf16x8;
typedef __attribute__((ext_vector_type(4)))  __bf16 bf16x4;
typedef __attribute__((ext_vector_type(8)))  float  f32x8;

#define NB   8
#define NA   1024
#define NNB  64      // neighbors per atom
#define NS   50      // spatial (rbf) basis
#define NSP  64      // NS padded to WMMA K granularity
#define NF   128     // filters == atom basis
#define CUT  5.0f

#if __has_builtin(__builtin_amdgcn_global_load_async_to_lds_b128)
#define HAVE_ASYNC_LDS 1
#else
#define HAVE_ASYNC_LDS 0
#endif

// Builtin signature (from hipcc diagnostic): param0 is
//   int __attribute__((vector_size(16))) addrspace(1)*   (global src)
// so we cast via integers to exactly-typed AS-qualified vector pointers.
typedef int v4i __attribute__((vector_size(16)));
typedef __attribute__((address_space(1))) v4i* as1_v4i_p;
typedef __attribute__((address_space(3))) v4i* as3_v4i_p;

// Async copy of one 16-byte chunk into LDS (per-lane addresses -> also works
// for gathers). Fallback: VGPR round-trip.
__device__ __forceinline__ void cp_async_b128(void* ldst, const void* gsrc) {
#if HAVE_ASYNC_LDS
    __builtin_amdgcn_global_load_async_to_lds_b128(
        (as1_v4i_p)(unsigned long long)gsrc,
        (as3_v4i_p)(unsigned)(unsigned long long)ldst,
        0, 0);
#else
    *(uint4*)ldst = *(const uint4*)gsrc;
#endif
}

__device__ __forceinline__ void async_wait0() {
#if HAVE_ASYNC_LDS
#if __has_builtin(__builtin_amdgcn_s_wait_asynccnt)
    __builtin_amdgcn_s_wait_asynccnt(0);
#else
    asm volatile("s_wait_asynccnt 0" ::: "memory");
#endif
#endif
}

__device__ __forceinline__ void cp16_async(void* dst, const void* src, int tid, int n16) {
    const uint4* s = (const uint4*)src;
    uint4*       d = (uint4*)dst;
    for (int i = tid; i < n16; i += 256) cp_async_b128(d + i, s + i);
}

__device__ __forceinline__ float sspf(float x) {
    // shifted softplus via hardware transcendentals (v_exp_f32 / v_log_f32):
    // max(x,0) + log(1 + exp(-|x|)) - ln2
    return fmaxf(x, 0.0f) + __logf(1.0f + __expf(-fabsf(x))) - 0.69314718055994531f;
}

// Per-lane WMMA 16-bit fragment load (A 16x32 row-major / B as [n][k]):
// lane = 16*half + r ; elements [0..7] = k0..k0+7, [8..15] = k0+16..k0+23,
// where k0 = kstep*32 + 8*half  (ISA 7.12.2 layout).
__device__ __forceinline__ bf16x16 load_frag(const __bf16* rowPtr, int k0) {
    bf16x8 lo = *(const bf16x8*)(rowPtr + k0);
    bf16x8 hi = *(const bf16x8*)(rowPtr + k0 + 16);
    bf16x16 f;
#pragma unroll
    for (int i = 0; i < 8; ++i) { f[i] = lo[i]; f[i + 8] = hi[i]; }
    return f;
}

// ---------------------------------------------------------------------------
// Weight preconversion: f32 -> bf16, [n][k] layout (k contiguous), fw1 padded
// K 50->64 with zeros (matches zero-padded f_ij panel).
// ---------------------------------------------------------------------------
__global__ void prep_weights_kernel(const float* __restrict__ in2f_w,
                                    const float* __restrict__ fw1,
                                    const float* __restrict__ fw2,
                                    const float* __restrict__ f2out_w,
                                    const float* __restrict__ dense_w,
                                    __bf16* __restrict__ in2fb,
                                    __bf16* __restrict__ fw1b,
                                    __bf16* __restrict__ fw2b,
                                    __bf16* __restrict__ f2outb,
                                    __bf16* __restrict__ denseb) {
    const int i = blockIdx.x;   // row (output-feature index), 0..127
    const int t = threadIdx.x;  // col, 0..127
    in2fb [i * NF + t] = (__bf16)in2f_w [i * NF + t];
    fw2b  [i * NF + t] = (__bf16)fw2    [i * NF + t];
    f2outb[i * NF + t] = (__bf16)f2out_w[i * NF + t];
    denseb[i * NF + t] = (__bf16)dense_w[i * NF + t];
    if (t < NSP) fw1b[i * NSP + t] = (__bf16)((t < NS) ? fw1[i * NS + t] : 0.0f);
}

// ---------------------------------------------------------------------------
// Generic [M x 128] (f32) x [128 x 128] (bf16, [n][k]) GEMM, optional bias and
// shifted-softplus epilogue. 64 rows per WG, 8 wave32s, 16 WMMA per wave.
// B panel staged with async-to-LDS (overlapped with A conversion).
// ---------------------------------------------------------------------------
template <int ACT>
__global__ __launch_bounds__(256) void gemm128_kernel(const float*  __restrict__ Am,
                                                      const __bf16* __restrict__ Bw,
                                                      const float*  __restrict__ bias,
                                                      float*        __restrict__ Out) {
    extern __shared__ char smem[];
    __bf16* aP = (__bf16*)smem;                  // [64][128] bf16, 16 KB
    __bf16* bP = (__bf16*)(smem + 64 * 128 * 2); // [128][128] bf16, 32 KB
    const int t    = threadIdx.x;
    const int row0 = blockIdx.x * 64;

    cp16_async(bP, Bw, t, 2048);  // 32 KB of bf16 weights, async into LDS

    {   // stage + convert A panel (overlaps with async B copy)
        const int row = t >> 2, q = t & 3;
        const float4* src = (const float4*)(Am + (size_t)(row0 + row) * NF + q * 32);
        __bf16*       dst = aP + row * NF + q * 32;
#pragma unroll
        for (int j = 0; j < 8; ++j) {
            float4 f = src[j];
            bf16x4 h; h[0] = (__bf16)f.x; h[1] = (__bf16)f.y; h[2] = (__bf16)f.z; h[3] = (__bf16)f.w;
            *(bf16x4*)(dst + j * 4) = h;
        }
    }
    async_wait0();
    __syncthreads();

    const int lane = t & 31, w = t >> 5;
    const int half = lane >> 4, r = lane & 15;
    const int mt  = w >> 1;          // m-tile 0..3
    const int nb0 = (w & 1) * 64;    // n range base
    f32x8 acc[4] = {};
    const __bf16* aRow = aP + (mt * 16 + r) * NF;
#pragma unroll
    for (int ks = 0; ks < 4; ++ks) {
        const int k0 = ks * 32 + 8 * half;
        bf16x16 af = load_frag(aRow, k0);
#pragma unroll
        for (int nt = 0; nt < 4; ++nt) {
            bf16x16 bfm = load_frag(bP + (nb0 + nt * 16 + r) * NF, k0);
            acc[nt] = __builtin_amdgcn_wmma_f32_16x16x32_bf16(
                false, af, false, bfm, (short)0, acc[nt], false, false);
        }
    }
#pragma unroll
    for (int nt = 0; nt < 4; ++nt) {
        const int n  = nb0 + nt * 16 + r;
        const float bv = bias ? bias[n] : 0.0f;
#pragma unroll
        for (int v = 0; v < 8; ++v) {
            const int m = mt * 16 + v + 8 * half;   // C layout: VGPR v -> M = v + 8*half
            float val = acc[nt][v] + bv;
            if (ACT) val = sspf(val);
            Out[(size_t)(row0 + m) * NF + n] = val;
        }
    }
}

// ---------------------------------------------------------------------------
// Fused per-atom kernel: filter net (GEMM1+ssp, GEMM2), cosine cutoff, neighbor
// gather, masked reduction -> agg[B*A][128].
// LDS: fij 8K | fw1 16K | W1 16K | fw2 32K | y_nbh 32K | scalars ~1.5K
// ---------------------------------------------------------------------------
#define SM_FIJ   0
#define SM_FW1   8192
#define SM_W1    24576
#define SM_FW2   40960
#define SM_YNB   73728
#define SM_CUT   106496
#define SM_MSK   106752
#define SM_NBR   107008
#define SM_AGG   107264
#define SM_MAIN  107776

__global__ __launch_bounds__(256) void schnet_main_kernel(
        const float* __restrict__ r_ij,   const int* __restrict__ neighbors,
        const float* __restrict__ nmask,  const float* __restrict__ f_ij,
        const float* __restrict__ y,
        const __bf16* __restrict__ fw1b,  const __bf16* __restrict__ fw2b,
        const float* __restrict__ fb1,    const float* __restrict__ fb2,
        float* __restrict__ aggOut) {
    extern __shared__ char smem[];
    __bf16* fij  = (__bf16*)(smem + SM_FIJ);   // [64][64]
    __bf16* fw1p = (__bf16*)(smem + SM_FW1);   // [128][64]
    __bf16* w1   = (__bf16*)(smem + SM_W1);    // [64][128]
    __bf16* fw2p = (__bf16*)(smem + SM_FW2);   // [128][128]
    float*  ynb  = (float*)(smem + SM_YNB);    // [64][128]
    float*  cutS = (float*)(smem + SM_CUT);
    float*  mskS = (float*)(smem + SM_MSK);
    int*    nbrS = (int*)  (smem + SM_NBR);
    float*  aggS = (float*)(smem + SM_AGG);

    const int t   = threadIdx.x;
    const int ba  = blockIdx.x;          // atom index in [0, B*A)
    const int b   = ba >> 10;
    const int base64 = ba * NNB;

    // ---- phase A: async weight panels, scalars, f_ij panel (bf16, padded) ----
    cp16_async(fw1p, fw1b, t, 1024);   // 16 KB
    cp16_async(fw2p, fw2b, t, 2048);   // 32 KB
    if (t < NNB) {
        const float rr = r_ij[base64 + t];
        cutS[t] = (rr < CUT) ? 0.5f * (__cosf(rr * 0.62831853071795865f) + 1.0f) : 0.0f;
        mskS[t] = nmask[base64 + t];
        nbrS[t] = neighbors[base64 + t];
    }
    if (t < NF) aggS[t] = 0.0f;
    {
        const int row = t >> 2, q = t & 3;
        const float* src = f_ij + (size_t)(base64 + row) * NS;
#pragma unroll
        for (int j = 0; j < 16; ++j) {
            const int c = q * 16 + j;
            fij[row * NSP + c] = (__bf16)((c < NS) ? src[c] : 0.0f);
        }
    }
    async_wait0();
    __syncthreads();

    // ---- gather neighbor features y[b, nbr[m], :] into LDS (async, per-lane addrs) ----
    {
        const int m = t >> 2, q = t & 3;
        const int g = b * NA + nbrS[m];
        const uint4* src = (const uint4*)(y + (size_t)g * NF) + q * 8;
        uint4*       dst = (uint4*)(ynb + m * NF) + q * 8;
#pragma unroll
        for (int j = 0; j < 8; ++j) cp_async_b128(dst + j, src + j);
    }
    async_wait0();
    __syncthreads();

    const int lane = t & 31, w = t >> 5;
    const int half = lane >> 4, r = lane & 15;
    const int mt  = w >> 1;
    const int nb0 = (w & 1) * 64;

    // ---- GEMM1: [64 x 64pad] x [64pad x 128] + fb1, ssp -> W1 (bf16 LDS) ----
    {
        f32x8 acc[4] = {};
        const __bf16* aRow = fij + (mt * 16 + r) * NSP;
#pragma unroll
        for (int ks = 0; ks < 2; ++ks) {
            const int k0 = ks * 32 + 8 * half;
            bf16x16 af = load_frag(aRow, k0);
#pragma unroll
            for (int nt = 0; nt < 4; ++nt) {
                bf16x16 bfm = load_frag(fw1p + (nb0 + nt * 16 + r) * NSP, k0);
                acc[nt] = __builtin_amdgcn_wmma_f32_16x16x32_bf16(
                    false, af, false, bfm, (short)0, acc[nt], false, false);
            }
        }
#pragma unroll
        for (int nt = 0; nt < 4; ++nt) {
            const int n  = nb0 + nt * 16 + r;
            const float b1 = fb1[n];
#pragma unroll
            for (int v = 0; v < 8; ++v) {
                const int m = mt * 16 + v + 8 * half;
                w1[m * NF + n] = (__bf16)sspf(acc[nt][v] + b1);
            }
        }
    }
    __syncthreads();

    // ---- GEMM2 + cutoff*mask*y_nbh reduction into agg ----
    {
        f32x8 acc[4] = {};
        const __bf16* aRow = w1 + (mt * 16 + r) * NF;
#pragma unroll
        for (int ks = 0; ks < 4; ++ks) {
            const int k0 = ks * 32 + 8 * half;
            bf16x16 af = load_frag(aRow, k0);
#pragma unroll
            for (int nt = 0; nt < 4; ++nt) {
                bf16x16 bfm = load_frag(fw2p + (nb0 + nt * 16 + r) * NF, k0);
                acc[nt] = __builtin_amdgcn_wmma_f32_16x16x32_bf16(
                    false, af, false, bfm, (short)0, acc[nt], false, false);
            }
        }
#pragma unroll
        for (int nt = 0; nt < 4; ++nt) {
            const int n  = nb0 + nt * 16 + r;
            const float b2 = fb2[n];
            float partial = 0.0f;
#pragma unroll
            for (int v = 0; v < 8; ++v) {
                const int m = mt * 16 + v + 8 * half;   // m = neighbor slot
                const float s = cutS[m] * mskS[m];
                partial += (acc[nt][v] + b2) * s * ynb[m * NF + n];
            }
            atomicAdd(&aggS[n], partial);   // ds_add_f32; lanes L/L+16 & 4 waves share n
        }
    }
    __syncthreads();
    if (t < NF) aggOut[(size_t)ba * NF + t] = aggS[t];
}

// ---------------------------------------------------------------------------
// Host-side launcher
// ---------------------------------------------------------------------------
extern "C" void kernel_launch(void* const* d_in, const int* in_sizes, int n_in,
                              void* d_out, int out_size, void* d_ws, size_t ws_size,
                              hipStream_t stream) {
    (void)in_sizes; (void)n_in; (void)out_size; (void)ws_size;
    const float* x       = (const float*)d_in[0];
    const float* r_ij    = (const float*)d_in[1];
    const int*   nbrs    = (const int*)  d_in[2];
    const float* nmask   = (const float*)d_in[3];
    const float* f_ij    = (const float*)d_in[4];
    const float* in2f_w  = (const float*)d_in[5];
    const float* fw1     = (const float*)d_in[6];
    const float* fb1     = (const float*)d_in[7];
    const float* fw2     = (const float*)d_in[8];
    const float* fb2     = (const float*)d_in[9];
    const float* f2out_w = (const float*)d_in[10];
    const float* f2out_b = (const float*)d_in[11];
    const float* dense_w = (const float*)d_in[12];
    const float* dense_b = (const float*)d_in[13];

    char* ws = (char*)d_ws;
    float*  yBuf   = (float*)ws;                         // [8192][128] f32 (reused as v)
    float*  aggBuf = (float*)(ws + (4u << 20));          // [8192][128] f32
    __bf16* in2fb  = (__bf16*)(ws + (8u << 20));
    __bf16* fw1b   = in2fb  + 128 * 128;
    __bf16* fw2b   = fw1b   + 128 * 64;
    __bf16* f2outb = fw2b   + 128 * 128;
    __bf16* denseb = f2outb + 128 * 128;

    prep_weights_kernel<<<128, 128, 0, stream>>>(in2f_w, fw1, fw2, f2out_w, dense_w,
                                                 in2fb, fw1b, fw2b, f2outb, denseb);
    // y = x @ in2f_w^T  (bias-free)
    gemm128_kernel<0><<<128, 256, 49152, stream>>>(x, in2fb, nullptr, yBuf);
    // fused filter net + cutoff + gather + masked reduce
    schnet_main_kernel<<<NB * NA, 256, SM_MAIN, stream>>>(r_ij, nbrs, nmask, f_ij, yBuf,
                                                          fw1b, fw2b, fb1, fb2, aggBuf);
    // v = ssp(agg @ f2out_w^T + b)   (yBuf dead -> reuse as v)
    gemm128_kernel<1><<<128, 256, 49152, stream>>>(aggBuf, f2outb, f2out_b, yBuf);
    // out = v @ dense_w^T + b
    gemm128_kernel<0><<<128, 256, 49152, stream>>>(yBuf, denseb, dense_b, (float*)d_out);
}
